// ABCNN_1_28200755265958
// MI455X (gfx1250) — matrix operand order repaired and checked by reading, not compiled, not run on hardware
//
#include <hip/hip_runtime.h>
#include <hip/hip_bf16.h>
#include <math.h>

typedef __attribute__((ext_vector_type(16))) _Float16 v16h;
typedef __attribute__((ext_vector_type(8)))  _Float16 v8h;
typedef __attribute__((ext_vector_type(8)))  float    v8f;

#define BATCH 64
#define SEQ   256
#define DIM   512
#define CWIDTH 3

// ---------------------------------------------------------------------------
// Kernel 1: F = Fr * mask  -> f16 copy + row sum-of-squares (f32)
// grid: (BATCH*SEQ, 2), block: 256
// ---------------------------------------------------------------------------
__global__ void prep_rows_kernel(const float* __restrict__ F0r,
                                 const float* __restrict__ F1r,
                                 const float* __restrict__ m0,
                                 const float* __restrict__ m1,
                                 _Float16* __restrict__ F0h,
                                 _Float16* __restrict__ F1h,
                                 float* __restrict__ sq0,
                                 float* __restrict__ sq1) {
  const int row   = blockIdx.x;            // b*SEQ + s
  const int which = blockIdx.y;
  const float* Fr = which ? F1r : F0r;
  const float* mk = which ? m1  : m0;
  _Float16*   Fh  = which ? F1h : F0h;
  float*      sq  = which ? sq1 : sq0;

  const float m = mk[row];
  const int   t = threadIdx.x;
  float acc = 0.f;
#pragma unroll
  for (int e = 0; e < DIM / 256; ++e) {
    const int d = t + e * 256;
    const float v = Fr[(size_t)row * DIM + d] * m;
    acc += v * v;
    Fh[(size_t)row * DIM + d] = (_Float16)v;
  }
  __shared__ float red[256];
  red[t] = acc;
  __syncthreads();
  for (int off = 128; off > 0; off >>= 1) {
    if (t < off) red[t] += red[t + off];
    __syncthreads();
  }
  if (t == 0) sq[row] = red[0];
}

// ---------------------------------------------------------------------------
// Kernel 2: transpose W [S,D] f32 -> Wt [D,S] f16
// grid: (S*D/256, 2), block: 256
// ---------------------------------------------------------------------------
__global__ void prep_wt_kernel(const float* __restrict__ W0,
                               const float* __restrict__ W1,
                               _Float16* __restrict__ W0th,
                               _Float16* __restrict__ W1th) {
  const int which = blockIdx.y;
  const float* W  = which ? W1   : W0;
  _Float16*   Wt  = which ? W1th : W0th;
  const int idx = blockIdx.x * blockDim.x + threadIdx.x;   // over S*D
  const int s = idx / DIM;
  const int d = idx % DIM;
  Wt[(size_t)d * SEQ + s] = (_Float16)W[idx];
}

// Build a 16x32 f16 A-fragment (contiguous k: lane half h holds k = 16h + e) from a row base.
__device__ __forceinline__ v16h load_afrag(const _Float16* rowk, int aoff) {
  const v8h lo = *(const v8h*)(rowk + aoff);
  const v8h hi = *(const v8h*)(rowk + 8 + aoff);
  v16h a;
#pragma unroll
  for (int e = 0; e < 8; ++e) { a[e] = lo[e]; a[8 + e] = hi[e]; }
  return a;
}

// ---------------------------------------------------------------------------
// Kernel 3: cross = F0 @ F1^T  ->  A = 1/(1+sqrt(max(sq0+sq1-2c,0)))
// One wave per 32x32 tile of A (2x2 register blocking -> 4 WMMA per K-step).
// grid: BATCH*8*8/4 blocks of 128 (4 waves)
// ---------------------------------------------------------------------------
__global__ void cross_kernel(const _Float16* __restrict__ F0h,
                             const _Float16* __restrict__ F1h,
                             const float* __restrict__ sq0,
                             const float* __restrict__ sq1,
                             _Float16* __restrict__ Ah,
                             _Float16* __restrict__ Ath) {
  const int lane = threadIdx.x & 31;
  const int wave = threadIdx.x >> 5;
  const int tile = blockIdx.x * (blockDim.x >> 5) + wave;  // 0..4095
  const int b  = tile >> 6;                                // 64 tiles / batch
  const int ti = (tile >> 3) & 7;
  const int tj = tile & 7;
  const int i0 = ti * 32, j0 = tj * 32;
  const int ml   = lane & 15;
  const int hi16 = lane >> 4;
  const int aoff = hi16 * 16;   // A frag: lanes 16-31 take K+16
  const int boff = hi16 * 16;   // B frag: lanes 16-31 take K+16

  const _Float16* arow0 = F0h + ((size_t)(b * SEQ + i0 + ml)) * DIM;
  const _Float16* arow1 = arow0 + (size_t)16 * DIM;
  const _Float16* brow0 = F1h + ((size_t)(b * SEQ + j0 + ml)) * DIM;
  const _Float16* brow1 = brow0 + (size_t)16 * DIM;

  v8f acc00 = {}, acc01 = {}, acc10 = {}, acc11 = {};
  for (int k = 0; k < DIM; k += 32) {
    const v16h a0 = load_afrag(arow0 + k, aoff);
    const v16h a1 = load_afrag(arow1 + k, aoff);
    const v16h b0 = *(const v16h*)(brow0 + k + boff);
    const v16h b1 = *(const v16h*)(brow1 + k + boff);
    acc00 = __builtin_amdgcn_wmma_f32_16x16x32_f16(false, a0, false, b0,
                                                   (short)0, acc00, false, false);
    acc01 = __builtin_amdgcn_wmma_f32_16x16x32_f16(false, a0, false, b1,
                                                   (short)0, acc01, false, false);
    acc10 = __builtin_amdgcn_wmma_f32_16x16x32_f16(false, a1, false, b0,
                                                   (short)0, acc10, false, false);
    acc11 = __builtin_amdgcn_wmma_f32_16x16x32_f16(false, a1, false, b1,
                                                   (short)0, acc11, false, false);
  }

  const v8f* accs[4] = {&acc00, &acc01, &acc10, &acc11};
#pragma unroll
  for (int f = 0; f < 4; ++f) {
    const int mi = f >> 1, ni = f & 1;
    const v8f acc = *accs[f];
#pragma unroll
    for (int r = 0; r < 8; ++r) {
      const int i = i0 + mi * 16 + r + hi16 * 8;  // C/D layout: M = r (+8 upper)
      const int j = j0 + ni * 16 + ml;            // N = lane & 15
      const float c  = acc[r];
      const float d2 = fmaxf(sq0[b * SEQ + i] + sq1[b * SEQ + j] - 2.0f * c, 0.0f);
      const float av = 1.0f / (1.0f + sqrtf(d2));
      const _Float16 ah = (_Float16)av;
      Ah [((size_t)(b * SEQ + i)) * SEQ + j] = ah;
      Ath[((size_t)(b * SEQ + j)) * SEQ + i] = ah;
    }
  }
}

// ---------------------------------------------------------------------------
// Kernel 4: F1a = A @ W1 (sel 0), F0a = A^T @ W0 (sel 1), f16 outputs.
// One wave per 32x32 output tile (2x2 blocking); K loop over SEQ.
// grid: 2*BATCH*8*16/4 blocks of 128
// ---------------------------------------------------------------------------
__global__ void apply_kernel(const _Float16* __restrict__ Ah,
                             const _Float16* __restrict__ Ath,
                             const _Float16* __restrict__ W0th,
                             const _Float16* __restrict__ W1th,
                             _Float16* __restrict__ F0a,
                             _Float16* __restrict__ F1a) {
  const int lane = threadIdx.x & 31;
  const int wave = threadIdx.x >> 5;
  int tile = blockIdx.x * (blockDim.x >> 5) + wave;        // 0..16383
  const int sel = tile >> 13;                              // uniform per wave
  tile &= 8191;
  const int b  = tile >> 7;                                // 128 tiles / batch
  const int tm = (tile >> 4) & 7;                          // SEQ/32
  const int tn = tile & 15;                                // DIM/32
  const int m0 = tm * 32, n0 = tn * 32;
  const int ml   = lane & 15;
  const int hi16 = lane >> 4;
  const int aoff = hi16 * 16;
  const int boff = hi16 * 16;

  const _Float16* Amat = sel ? Ath  : Ah;
  const _Float16* Wt   = sel ? W0th : W1th;
  _Float16*       Out  = sel ? F0a  : F1a;

  const _Float16* arow0 = Amat + ((size_t)(b * SEQ + m0 + ml)) * SEQ;
  const _Float16* arow1 = arow0 + (size_t)16 * SEQ;
  const _Float16* bcol0 = Wt + ((size_t)(n0 + ml)) * SEQ;   // [D,S]: K contiguous
  const _Float16* bcol1 = bcol0 + (size_t)16 * SEQ;

  v8f acc00 = {}, acc01 = {}, acc10 = {}, acc11 = {};
#pragma unroll
  for (int k = 0; k < SEQ; k += 32) {
    const v16h a0 = load_afrag(arow0 + k, aoff);
    const v16h a1 = load_afrag(arow1 + k, aoff);
    const v16h b0 = *(const v16h*)(bcol0 + k + boff);
    const v16h b1 = *(const v16h*)(bcol1 + k + boff);
    acc00 = __builtin_amdgcn_wmma_f32_16x16x32_f16(false, a0, false, b0,
                                                   (short)0, acc00, false, false);
    acc01 = __builtin_amdgcn_wmma_f32_16x16x32_f16(false, a0, false, b1,
                                                   (short)0, acc01, false, false);
    acc10 = __builtin_amdgcn_wmma_f32_16x16x32_f16(false, a1, false, b0,
                                                   (short)0, acc10, false, false);
    acc11 = __builtin_amdgcn_wmma_f32_16x16x32_f16(false, a1, false, b1,
                                                   (short)0, acc11, false, false);
  }

  const v8f* accs[4] = {&acc00, &acc01, &acc10, &acc11};
#pragma unroll
  for (int f = 0; f < 4; ++f) {
    const int mi = f >> 1, ni = f & 1;
    const v8f acc = *accs[f];
#pragma unroll
    for (int r = 0; r < 8; ++r) {
      const int m = m0 + mi * 16 + r + hi16 * 8;
      const int n = n0 + ni * 16 + ml;
      Out[((size_t)(b * SEQ + m)) * DIM + n] = (_Float16)acc[r];
    }
  }
}

// ---------------------------------------------------------------------------
// Kernel 5: conv (2ch, height-3, pad 2) + bias + tanh + avgpool-3 -> out
// thread per (b,s,d); grid: (B*S*D/256, 2), block 256
// ---------------------------------------------------------------------------
__global__ void conv_kernel(const float* __restrict__ F0r,
                            const float* __restrict__ F1r,
                            const float* __restrict__ m0,
                            const float* __restrict__ m1,
                            const _Float16* __restrict__ F0a,
                            const _Float16* __restrict__ F1a,
                            const float* __restrict__ conv_w,
                            const float* __restrict__ conv_b,
                            float* __restrict__ out) {
  const int which = blockIdx.y;
  const float* Fr = which ? F1r : F0r;
  const float* mk = which ? m1  : m0;
  const _Float16* Fa = which ? F1a : F0a;
  float* o = out + (size_t)which * BATCH * SEQ * DIM;

  const int tid = blockIdx.x * blockDim.x + threadIdx.x;
  const int d = tid & (DIM - 1);
  const int s = (tid >> 9) & (SEQ - 1);
  const int b = tid >> 17;

  float w0[CWIDTH], w1[CWIDTH];
#pragma unroll
  for (int kh = 0; kh < CWIDTH; ++kh) {
    w0[kh] = conv_w[kh];             // channel 0: masked F
    w1[kh] = conv_w[CWIDTH + kh];    // channel 1: attention feature
  }
  const float bias = conv_b[0];

  float xc[5], xa[5];                // rows s-2 .. s+2 (zero-padded)
#pragma unroll
  for (int u = 0; u < 5; ++u) {
    const int row = s + u - 2;
    const bool ok = (row >= 0) && (row < SEQ);
    const int rr = ok ? row : 0;
    const size_t idx = ((size_t)(b * SEQ + rr)) * DIM + d;
    xc[u] = ok ? Fr[idx] * mk[b * SEQ + rr] : 0.0f;
    xa[u] = ok ? (float)Fa[idx] : 0.0f;
  }

  float z = 0.0f;
#pragma unroll
  for (int t = 0; t < 3; ++t) {      // pool window over conv outputs s..s+2
    float y = bias;
#pragma unroll
    for (int kh = 0; kh < CWIDTH; ++kh)
      y += w0[kh] * xc[t + kh] + w1[kh] * xa[t + kh];
    z += tanhf(y);
  }
  o[tid] = z * (1.0f / 3.0f);
}

// ---------------------------------------------------------------------------
extern "C" void kernel_launch(void* const* d_in, const int* in_sizes, int n_in,
                              void* d_out, int out_size, void* d_ws, size_t ws_size,
                              hipStream_t stream) {
  const float* F0r = (const float*)d_in[0];
  const float* F1r = (const float*)d_in[1];
  const float* m0  = (const float*)d_in[2];
  const float* m1  = (const float*)d_in[3];
  const float* W0  = (const float*)d_in[4];
  const float* W1  = (const float*)d_in[5];
  const float* cw  = (const float*)d_in[6];
  const float* cb  = (const float*)d_in[7];
  float* out = (float*)d_out;

  const size_t nBSD = (size_t)BATCH * SEQ * DIM;   // 8,388,608
  const size_t nBSS = (size_t)BATCH * SEQ * SEQ;   // 4,194,304
  const size_t nSD  = (size_t)SEQ * DIM;
  const size_t nBS  = (size_t)BATCH * SEQ;

  char* p = (char*)d_ws;
  _Float16* F0h  = (_Float16*)p; p += nBSD * sizeof(_Float16);
  _Float16* F1h  = (_Float16*)p; p += nBSD * sizeof(_Float16);
  _Float16* W0th = (_Float16*)p; p += nSD  * sizeof(_Float16);
  _Float16* W1th = (_Float16*)p; p += nSD  * sizeof(_Float16);
  _Float16* Ah   = (_Float16*)p; p += nBSS * sizeof(_Float16);
  _Float16* Ath  = (_Float16*)p; p += nBSS * sizeof(_Float16);
  _Float16* F0a  = (_Float16*)p; p += nBSD * sizeof(_Float16);
  _Float16* F1a  = (_Float16*)p; p += nBSD * sizeof(_Float16);
  float*    sq0  = (float*)p;    p += nBS  * sizeof(float);
  float*    sq1  = (float*)p;    p += nBS  * sizeof(float);

  // 1) mask + f16 convert + row sumsq
  prep_rows_kernel<<<dim3(BATCH * SEQ, 2), 256, 0, stream>>>(
      F0r, F1r, m0, m1, F0h, F1h, sq0, sq1);

  // 2) weight transpose to f16 [D,S]
  prep_wt_kernel<<<dim3((SEQ * DIM) / 256, 2), 256, 0, stream>>>(
      W0, W1, W0th, W1th);

  // 3) cross-similarity -> A and A^T (f16), 32x32 per wave
  cross_kernel<<<dim3((BATCH * 8 * 8) / 4), 128, 0, stream>>>(
      F0h, F1h, sq0, sq1, Ah, Ath);

  // 4) F1a = A @ W1 and F0a = A^T @ W0, 32x32 per wave, f16 out
  apply_kernel<<<dim3((2 * BATCH * 8 * 16) / 4), 128, 0, stream>>>(
      Ah, Ath, W0th, W1th, F0a, F1a);

  // 5) conv + tanh + pool for both outputs
  conv_kernel<<<dim3((BATCH * SEQ * DIM) / 256, 2), 256, 0, stream>>>(
      F0r, F1r, m0, m1, F0a, F1a, cw, cb, out);
}
